// DeltaNetEnhancedAttention_53077205844585
// MI455X (gfx1250) — matrix operand order repaired and checked by reading, not compiled, not run on hardware
//
#include <hip/hip_runtime.h>
#include <hip/hip_bf16.h>

typedef __attribute__((ext_vector_type(16))) __bf16 v16bf;
typedef __attribute__((ext_vector_type(8)))  float  v8f;
typedef __attribute__((ext_vector_type(4)))  unsigned int u32x4;
typedef __attribute__((ext_vector_type(8)))  int i32x8;
typedef __attribute__((ext_vector_type(4)))  int i32x4;

#if defined(__gfx1250__) && __has_builtin(__builtin_amdgcn_tensor_load_to_lds) && \
    __has_builtin(__builtin_amdgcn_s_wait_tensorcnt)
#define USE_TDM 1
#else
#define USE_TDM 0
#endif

// ---------- helpers ----------
__device__ __forceinline__ unsigned short f2bf(float f) {
  unsigned u = __float_as_uint(f);
  unsigned r = u + 0x7FFFu + ((u >> 16) & 1u);   // round-to-nearest-even
  return (unsigned short)(r >> 16);
}

#if USE_TDM
// Issue a 2D TDM tile load: rows x cols elements, row stride 'stride' (elements),
// element size by code ds2 (1=2B, 2=4B), from global 'ga' into LDS offset 'ldsa'.
__device__ __forceinline__ void tdm_load_2d(unsigned long long ga, unsigned ldsa,
                                            int cols, int rows, int stride, int ds2) {
  u32x4 g0;
  g0.x = 1u;                                   // count=1 (valid user descriptor)
  g0.y = ldsa;                                 // lds_addr (bytes)
  g0.z = (unsigned)ga;                         // global_addr[31:0]
  g0.w = (unsigned)(ga >> 32) | (2u << 30);    // global_addr[56:32] | type=2
  i32x8 g1;
  unsigned T0 = 1u << 20, T1 = 1u << 20;       // generous tensor dims (no OOB clip)
  g1[0] = ds2 << 16;                                         // data_size
  g1[1] = (int)((T0 & 0xFFFFu) << 16);                       // tdim0 lo16 @ [63:48]
  g1[2] = (int)((T0 >> 16) | ((T1 & 0xFFFFu) << 16));        // tdim0 hi | tdim1 lo
  g1[3] = (int)((T1 >> 16) | ((unsigned)cols << 16));        // tdim1 hi | tile_dim0
  g1[4] = rows;                                              // tile_dim1 (tile_dim2=0)
  g1[5] = stride;                                            // dim0 stride lo32
  g1[6] = 0;                                                 // stride hi, dim1 stride lo
  g1[7] = 0;
  i32x4 z4; z4[0] = 0; z4[1] = 0; z4[2] = 0; z4[3] = 0;
  i32x8 z8;
  for (int i = 0; i < 8; ++i) z8[i] = 0;
  // 6-arg form (clang-23 / therock-10.0 headers): groups 0..3 + extra group + cpol
  __builtin_amdgcn_tensor_load_to_lds(g0, g1, z4, z4, z8, 0);
}
#endif

// ---------- f32 -> bf16 convert ----------
__global__ void k_f32_to_bf16(const float* __restrict__ src,
                              unsigned short* __restrict__ dst, int n) {
  int i = blockIdx.x * blockDim.x + threadIdx.x;
  if (i < n) dst[i] = f2bf(src[i]);
}

// ---------- big GEMM: C(f32 MxN) = A(bf16 MxK) * B(bf16 KxN) [+bias][gelu] ----------
// block = 256 threads (8 waves), tile 64x64, each wave owns 2 16x16 WMMA tiles.
__global__ void k_wmma_gemm(const unsigned short* __restrict__ A,
                            const unsigned short* __restrict__ Bw,
                            const float* __restrict__ bias,
                            float* __restrict__ C,
                            int M, int N, int K, int epi) {
  __shared__ unsigned short As[64 * 32];   // [m][k], k contiguous
  __shared__ unsigned short Bs[64 * 32];   // [n][k], k contiguous (transposed on load)
  const int tid  = threadIdx.x;
  const int lane = tid & 31, wave = tid >> 5;
  const int m0 = blockIdx.y * 64, n0 = blockIdx.x * 64;
  const int mt = wave >> 1;                 // wave's M tile (0..3)
  const int mlane = lane & 15, kh = lane >> 4;

  v8f acc[2];
  for (int t = 0; t < 2; ++t)
    for (int r = 0; r < 8; ++r) acc[t][r] = 0.0f;

  unsigned*       AsU = (unsigned*)As;
  unsigned*       BsU = (unsigned*)Bs;
#if !USE_TDM
  const unsigned* Au  = (const unsigned*)A;
#endif

  for (int kk = 0; kk < K; kk += 32) {
    __syncthreads();
#if USE_TDM
    // A tile via Tensor Data Mover: 64 rows x 32 bf16, row stride K (elements)
    if (wave == 0) {
      unsigned long long ga = (unsigned long long)A +
          (((unsigned long long)m0 * (unsigned)K + (unsigned)kk) << 1);
      tdm_load_2d(ga, (unsigned)(unsigned long long)(void*)As, 32, 64, K, 1);
    }
#else
    for (int i = tid; i < 1024; i += 256) {
      int r = i >> 4, c = i & 15;
      AsU[i] = Au[(size_t)(m0 + r) * (K >> 1) + (kk >> 1) + c];
    }
#endif
    // stage B tile transposed: Bs[n][k] from B[k][n]
    for (int i = tid; i < 2048; i += 256) {
      int n = i >> 5, kl = i & 31;
      Bs[n * 32 + kl] = Bw[(size_t)(kk + kl) * N + n0 + n];
    }
    if (lane == 0 && kk + 32 < K)
      __builtin_prefetch(&Bw[(size_t)(kk + 32) * N + n0], 0, 3);
#if USE_TDM
    if (wave == 0) __builtin_amdgcn_s_wait_tensorcnt(0);
#endif
    __syncthreads();

    // A fragment (16x32 bf16): lane m = lane&15, K half by lane>>4,
    // VGPR v holds K pair {base, base+1}, base = (v<4 ? 2v : 16+2(v-4)) + 8*kh
    union { v16bf v; unsigned u[8]; } fa;
    const int am = mt * 16 + mlane;
    for (int vv = 0; vv < 8; ++vv) {
      int kb = (vv < 4 ? 2 * vv : 16 + 2 * (vv - 4)) + 8 * kh;
      fa.u[vv] = AsU[am * 16 + (kb >> 1)];
    }
    for (int t = 0; t < 2; ++t) {
      int nt = (wave * 2 + t) & 3;
      // B fragment (32x16 bf16): lane n = lane&15, VGPR v holds K pair, same layout as A
      union { v16bf v; unsigned u[8]; } fb;
      const int bn = nt * 16 + mlane;
      for (int vv = 0; vv < 8; ++vv)
        fb.u[vv] = BsU[bn * 16 + (vv < 4 ? vv : 4 + vv) + 4 * kh];
      acc[t] = __builtin_amdgcn_wmma_f32_16x16x32_bf16(
          false, fa.v, false, fb.v, (short)0, acc[t], false, false);
    }
  }

  for (int t = 0; t < 2; ++t) {
    int nt  = (wave * 2 + t) & 3;
    int col = n0 + nt * 16 + mlane;
    float bv = bias ? bias[col] : 0.0f;
    for (int r = 0; r < 8; ++r) {
      int row = m0 + mt * 16 + r + 8 * kh;
      float xv = acc[t][r] + bv;
      if (epi == 1) {  // tanh-approx GELU (jax.nn.gelu default)
        float x3 = xv * xv * xv;
        xv = 0.5f * xv * (1.0f + tanhf(0.7978845608028654f * (xv + 0.044715f * x3)));
      }
      C[(size_t)row * N + col] = xv;
    }
  }
}

// ---------- small GEMM (N in {16,80}): naive f32 ----------
__global__ void k_small_gemm(const float* __restrict__ X, const float* __restrict__ W,
                             const float* __restrict__ bias, float* __restrict__ Y,
                             int M, int N, int K) {
  int i = blockIdx.x * blockDim.x + threadIdx.x;
  if (i >= M * N) return;
  int m = i / N, n = i - m * N;
  float acc = bias ? bias[n] : 0.0f;
  for (int k = 0; k < K; ++k) acc += X[(size_t)m * K + k] * W[(size_t)k * N + n];
  Y[i] = acc;
}

// ---------- in-LDS 64x64x64 matmul via WMMA (f32 tiles, bf16 on the fly) ----------
// C[m][n] (+)= sum_k opA[m][k]*opB[k][n] ; opA/opB select transpose gathers.
__device__ __forceinline__ void mm64(float* __restrict__ C, const float* __restrict__ A,
                                     const float* __restrict__ B, int transA, int transB,
                                     int accum, int tid) {
  __syncthreads();
  const int lane = tid & 31, wave = tid >> 5;
  const int mt = wave >> 1;
  const int mlane = lane & 15, kh = lane >> 4;
  v8f acc[2];
  for (int t = 0; t < 2; ++t) {
    const int nt = (wave * 2 + t) & 3;
    for (int r = 0; r < 8; ++r)
      acc[t][r] = accum ? C[(mt * 16 + r + 8 * kh) * 64 + nt * 16 + mlane] : 0.0f;
  }
  for (int kk = 0; kk < 64; kk += 32) {
    union { v16bf v; unsigned short s[16]; } fa;
    const int am = mt * 16 + mlane;
    for (int j = 0; j < 16; ++j) {
      int vv = j >> 1, lo = j & 1;
      int kl = (vv < 4 ? 2 * vv : 16 + 2 * (vv - 4)) + 8 * kh + lo;
      int kg = kk + kl;
      float val = transA ? A[kg * 64 + am] : A[am * 64 + kg];
      fa.s[j] = f2bf(val);
    }
    for (int t = 0; t < 2; ++t) {
      const int nt = (wave * 2 + t) & 3;
      union { v16bf v; unsigned short s[16]; } fb;
      const int bn = nt * 16 + mlane;
      for (int j = 0; j < 16; ++j) {
        int kg = kk + (j < 8 ? j : 8 + j) + 8 * kh;
        float val = transB ? B[bn * 64 + kg] : B[kg * 64 + bn];
        fb.s[j] = f2bf(val);
      }
      acc[t] = __builtin_amdgcn_wmma_f32_16x16x32_bf16(
          false, fa.v, false, fb.v, (short)0, acc[t], false, false);
    }
  }
  __syncthreads();
  for (int t = 0; t < 2; ++t) {
    const int nt = (wave * 2 + t) & 3;
    for (int r = 0; r < 8; ++r)
      C[(mt * 16 + r + 8 * kh) * 64 + nt * 16 + mlane] = acc[t][r];
  }
  __syncthreads();
}

// ---------- delta-rule chunkwise recurrence: one block per (b,h) ----------
__global__ void k_delta(const float* __restrict__ qkv, float* __restrict__ dout) {
  extern __shared__ float sm[];
  float* S  = sm;           // 64x64 state
  float* Q  = S  + 4096;
  float* Kt = Q  + 4096;    // normalized k
  float* V  = Kt + 4096;    // v*beta, later o
  float* Wt = V  + 4096;    // w solve
  float* U  = Wt + 4096;    // u solve -> u_adj
  float* At = U  + 4096;    // A, then w@S scratch
  float* Tt = At + 4096;    // attn_loc
  float* betaA = Tt + 4096; // 64
  const int bh = blockIdx.x;
  const int b = bh >> 4, h = bh & 15;
  const int tid = threadIdx.x;
  const int wave = tid >> 5;
  (void)wave;

  for (int i = tid; i < 4096; i += 256) S[i] = 0.0f;

  for (int g = 0; g < 16; ++g) {
    const int base = g * 64;
#if USE_TDM
    // waves 0/1/2 DMA the q/k/v 64x64 f32 tiles (row stride 3072 elements)
    if (wave < 3) {
      float* dst = (wave == 0) ? Q : (wave == 1) ? Kt : V;
      unsigned long long ga = (unsigned long long)qkv +
          (((unsigned long long)(b * 1024 + base)) * 3072ull +
           (unsigned)(wave * 1024) + (unsigned)(h * 64)) * 4ull;
      tdm_load_2d(ga, (unsigned)(unsigned long long)(void*)dst, 64, 64, 3072, 2);
    }
#else
    for (int i = tid; i < 4096; i += 256) {
      int c = i >> 6, d = i & 63;
      size_t idx = (size_t)(b * 1024 + base + c) * 3072 + h * 64 + d;
      Q[i]  = qkv[idx];
      Kt[i] = qkv[idx + 1024];
      V[i]  = qkv[idx + 2048];
    }
#endif
    if (tid < 64) betaA[tid] = 1.0f - 0.9f * (float)(base + tid) / 1023.0f;
#if USE_TDM
    if (wave < 3) __builtin_amdgcn_s_wait_tensorcnt(0);
#endif
    __syncthreads();
    if (tid < 64) {               // l2norm(q) * d^-1/2
      float s = 1e-6f;
      for (int d = 0; d < 64; ++d) { float q = Q[tid * 64 + d]; s += q * q; }
      float r = rsqrtf(s) * 0.125f;
      for (int d = 0; d < 64; ++d) Q[tid * 64 + d] *= r;
    } else if (tid < 128) {       // l2norm(k)
      int t = tid - 64;
      float s = 1e-6f;
      for (int d = 0; d < 64; ++d) { float k = Kt[t * 64 + d]; s += k * k; }
      float r = rsqrtf(s);
      for (int d = 0; d < 64; ++d) Kt[t * 64 + d] *= r;
    }
    __syncthreads();
    for (int i = tid; i < 4096; i += 256) V[i] *= betaA[i >> 6];   // v *= beta

    mm64(At, Kt, Kt, 0, 1, 0, tid);   // K K^T
    mm64(Tt, Q,  Kt, 0, 1, 0, tid);   // Q K^T
    for (int i = tid; i < 4096; i += 256) {
      int c = i >> 6, e = i & 63;
      At[i] = (e < c) ? At[i] * betaA[c] : 0.0f;   // A = tril(beta*KK^T, -1)
      if (e > c) Tt[i] = 0.0f;                     // attn = tril(QK^T)
    }
    for (int i = tid; i < 4096; i += 256) {        // init solves
      int c = i >> 6;
      Wt[i] = betaA[c] * Kt[i];                    // rhs kb
      U[i]  = V[i];                                // rhs v*beta
    }
    __syncthreads();
    // forward substitution (M = I + A unit-lower): rows sequential
    for (int r = 1; r < 64; ++r) {
      if (tid < 64) {
        float a = 0.0f;
        for (int j = 0; j < r; ++j) a += At[r * 64 + j] * Wt[j * 64 + tid];
        Wt[r * 64 + tid] -= a;
      } else if (tid < 128) {
        int d = tid - 64;
        float a = 0.0f;
        for (int j = 0; j < r; ++j) a += At[r * 64 + j] * U[j * 64 + d];
        U[r * 64 + d] -= a;
      }
      __syncthreads();
    }
    mm64(At, Wt, S, 0, 0, 0, tid);                         // w @ S
    for (int i = tid; i < 4096; i += 256) U[i] -= At[i];   // u_adj
    mm64(V, Q,  S, 0, 0, 0, tid);                          // o = q@S
    mm64(V, Tt, U, 0, 0, 1, tid);                          //   + attn@u_adj
    mm64(S, Kt, U, 1, 0, 1, tid);                          // S += k^T @ u_adj
    for (int i = tid; i < 4096; i += 256) {
      int c = i >> 6, d = i & 63;
      dout[((size_t)(b * 16 + h) * 1024 + base + c) * 64 + d] = V[i];
    }
    __syncthreads();
  }
}

// ---------- zero entropy slot ----------
__global__ void k_zero_one(float* p) { if (threadIdx.x == 0 && blockIdx.x == 0) p[0] = 0.0f; }

// ---------- fusion: FIR, cross-mix, gates, softmax, entropy, ctx(bf16) ----------
__global__ void k_fuse(const float* __restrict__ qkv, const float* __restrict__ idbuf,
                       const float* __restrict__ deltabuf, const float* __restrict__ gatebuf,
                       const float* __restrict__ prunebuf, const float* __restrict__ g2buf,
                       const float* __restrict__ fir_s, const float* __restrict__ fir_l,
                       const float* __restrict__ mix, const float* __restrict__ id_static,
                       unsigned short* __restrict__ ctx_bf, float* __restrict__ ent_out) {
  __shared__ float fs0[1024], fl0[1024], wg[80], idg[16], entacc;
  const int bn = blockIdx.x;
  const int b = bn >> 10, n = bn & 1023;
  const int tid = threadIdx.x;
  if (tid == 0) entacc = 0.0f;
  __syncthreads();
  // causal depthwise FIR on V path
  for (int e = tid; e < 1024; e += 256) {
    int h = e >> 6, d = e & 63;
    float s3 = 0.0f;
    for (int j = 0; j < 3; ++j) {
      int t = n - 2 + j;
      if (t >= 0) s3 += fir_s[(h * 64 + d) * 3 + j] * qkv[(size_t)(b * 1024 + t) * 3072 + 2048 + e];
    }
    fs0[e] = s3;
    float s7 = 0.0f;
    for (int j = 0; j < 7; ++j) {
      int t = n - 6 + j;
      if (t >= 0) s7 += fir_l[(h * 64 + d) * 7 + j] * qkv[(size_t)(b * 1024 + t) * 3072 + 2048 + e];
    }
    fl0[e] = s7;
  }
  if (tid < 16) {
    const int h = tid;
    float lg[5], mx = -1e30f;
    for (int p = 0; p < 5; ++p) { lg[p] = prunebuf[(size_t)bn * 80 + h * 5 + p]; mx = fmaxf(mx, lg[p]); }
    float den = 0.0f;
    for (int p = 0; p < 5; ++p) { lg[p] = __expf(lg[p] - mx); den += lg[p]; }
    float e1 = 0.0f, l2[5];
    for (int p = 0; p < 5; ++p) {
      float pr = lg[p] / den;
      e1 += pr * __logf(pr + 1e-8f);
      l2[p] = g2buf[(size_t)bn * 80 + h * 5 + p] + __logf(pr + 1e-8f);
    }
    mx = -1e30f;
    for (int p = 0; p < 5; ++p) mx = fmaxf(mx, l2[p]);
    den = 0.0f;
    for (int p = 0; p < 5; ++p) { l2[p] = __expf(l2[p] - mx); den += l2[p]; }
    float e2 = 0.0f;
    for (int p = 0; p < 5; ++p) {
      float w = l2[p] / den;
      wg[h * 5 + p] = w;
      e2 += w * __logf(w + 1e-8f);
    }
    atomicAdd(&entacc, e1 + e2);
    float dg = 1.0f / (1.0f + __expf(-gatebuf[(size_t)bn * 16 + h]));
    float sg = 1.0f / (1.0f + __expf(-id_static[h]));
    idg[h] = dg * sg;
  }
  __syncthreads();
  if (tid == 0) atomicAdd(ent_out, -entacc / 65536.0f);  // / (B*N*H), for both ent1+ent2
  for (int e = tid; e < 1024; e += 256) {
    int h = e >> 6, d = e & 63;
    float fsm = fs0[e], flm = fl0[e];
    for (int g = 0; g < 16; ++g) {
      float mg = mix[g * 16 + h];
      fsm += fs0[g * 64 + d] * mg;
      flm += fl0[g * 64 + d] * mg;
    }
    float dv = deltabuf[((size_t)(b * 16 + h) * 1024 + n) * 64 + d];
    float vv = qkv[(size_t)bn * 3072 + 2048 + e];
    float iv = idbuf[(size_t)bn * 1024 + e] * idg[h];
    const float* w = &wg[h * 5];
    float ctx = w[0] * fsm + w[1] * flm + w[2] * dv + w[3] * vv + w[4] * iv;
    ctx_bf[(size_t)bn * 1024 + e] = f2bf(ctx);
  }
}

// ---------- launch ----------
extern "C" void kernel_launch(void* const* d_in, const int* in_sizes, int n_in,
                              void* d_out, int out_size, void* d_ws, size_t ws_size,
                              hipStream_t stream) {
  (void)in_sizes; (void)n_in; (void)out_size; (void)ws_size;
  const float* x          = (const float*)d_in[0];
  const float* W_qkv      = (const float*)d_in[1];
  const float* W_id       = (const float*)d_in[2];
  const float* W_gate     = (const float*)d_in[3];
  const float* b_gate     = (const float*)d_in[4];
  const float* id_static  = (const float*)d_in[5];
  const float* fir_s      = (const float*)d_in[6];
  const float* fir_l      = (const float*)d_in[7];
  const float* mix        = (const float*)d_in[8];
  const float* W_prune    = (const float*)d_in[9];
  const float* b_prune    = (const float*)d_in[10];
  const float* W_g1       = (const float*)d_in[11];
  const float* b_g1       = (const float*)d_in[12];
  const float* W_g2       = (const float*)d_in[13];
  const float* b_g2       = (const float*)d_in[14];
  const float* W_proj     = (const float*)d_in[15];
  const float* b_proj     = (const float*)d_in[16];
  float* out = (float*)d_out;

  char* ws = (char*)d_ws;
  size_t off = 0;
  auto carve = [&](size_t bytes) -> void* {
    void* p = (void*)(ws + off);
    off += (bytes + 255) & ~(size_t)255;
    return p;
  };
  unsigned short* xb      = (unsigned short*)carve(4194304ull * 2);
  unsigned short* wqkvb   = (unsigned short*)carve(3145728ull * 2);
  unsigned short* widb    = (unsigned short*)carve(1048576ull * 2);
  unsigned short* wg1b    = (unsigned short*)carve(1048576ull * 2);
  unsigned short* wprojb  = (unsigned short*)carve(1048576ull * 2);
  float* qkvbuf   = (float*)carve(12582912ull * 4);
  float* idbuf    = (float*)carve(4194304ull * 4);
  float* hbuf     = (float*)carve(4194304ull * 4);
  float* gatebuf  = (float*)carve(65536ull * 4);
  float* prunebuf = (float*)carve(327680ull * 4);
  float* g2buf    = (float*)carve(327680ull * 4);
  float* deltabuf = (float*)carve(4194304ull * 4);
  unsigned short* ctxbf = (unsigned short*)carve(4194304ull * 2);

  k_f32_to_bf16<<<(4194304 + 255) / 256, 256, 0, stream>>>(x, xb, 4194304);
  k_f32_to_bf16<<<(3145728 + 255) / 256, 256, 0, stream>>>(W_qkv, wqkvb, 3145728);
  k_f32_to_bf16<<<(1048576 + 255) / 256, 256, 0, stream>>>(W_id, widb, 1048576);
  k_f32_to_bf16<<<(1048576 + 255) / 256, 256, 0, stream>>>(W_g1, wg1b, 1048576);
  k_f32_to_bf16<<<(1048576 + 255) / 256, 256, 0, stream>>>(W_proj, wprojb, 1048576);

  k_wmma_gemm<<<dim3(3072 / 64, 4096 / 64), 256, 0, stream>>>(xb, wqkvb, nullptr, qkvbuf, 4096, 3072, 1024, 0);
  k_wmma_gemm<<<dim3(1024 / 64, 4096 / 64), 256, 0, stream>>>(xb, widb,  nullptr, idbuf,  4096, 1024, 1024, 0);
  k_wmma_gemm<<<dim3(1024 / 64, 4096 / 64), 256, 0, stream>>>(xb, wg1b,  b_g1,    hbuf,   4096, 1024, 1024, 1);

  k_small_gemm<<<(65536  + 255) / 256, 256, 0, stream>>>(x,    W_gate,  b_gate,  gatebuf,  4096, 16, 1024);
  k_small_gemm<<<(327680 + 255) / 256, 256, 0, stream>>>(x,    W_prune, b_prune, prunebuf, 4096, 80, 1024);
  k_small_gemm<<<(327680 + 255) / 256, 256, 0, stream>>>(hbuf, W_g2,    b_g2,    g2buf,    4096, 80, 1024);

  k_delta<<<64, 256, 8 * 4096 * 4 + 256, stream>>>(qkvbuf, deltabuf);

  k_zero_one<<<1, 1, 0, stream>>>(out + 4194304);
  k_fuse<<<4096, 256, 0, stream>>>(qkvbuf, idbuf, deltabuf, gatebuf, prunebuf, g2buf,
                                   fir_s, fir_l, mix, id_static, ctxbf, out + 4194304);

  k_wmma_gemm<<<dim3(1024 / 64, 4096 / 64), 256, 0, stream>>>(ctxbf, wprojb, b_proj, out, 4096, 1024, 1024, 0);
}
